// CausalSelfAttention_27719718928587
// MI455X (gfx1250) — matrix operand order repaired and checked by reading, not compiled, not run on hardware
//
#include <hip/hip_runtime.h>

typedef __attribute__((ext_vector_type(16))) _Float16 v16h;
typedef __attribute__((ext_vector_type(8)))  _Float16 v8h;
typedef __attribute__((ext_vector_type(8)))  float    v8f;

#define BB    2
#define TT    2048
#define CC    1024
#define HH    16
#define HD    64
#define NQKV  3072
#define MROWS 4096      /* B*T */
#define ATT_SCALE 0.125f /* 1/sqrt(64) */

// ---------------------------------------------------------------------------
// WMMA helper: D = A(16x32 f16) x B(32x16 f16) + C(16x16 f32)
// ---------------------------------------------------------------------------
__device__ __forceinline__ v8f wmma32(v16h a, v16h b, v8f c) {
  return __builtin_amdgcn_wmma_f32_16x16x32_f16(
      /*neg_a=*/false, a, /*neg_b=*/false, b,
      /*c_mod=*/(short)0, c, /*reuse_a=*/false, /*reuse_b=*/false);
}

// A operand (16 rows x 32 contraction) from row-major [row][col], stride ld.
// ISA layout (16-bit A 16x32): lane<16: M=lane, K = {0..7, 16..23};
//                              lane>=16: M=lane-16, K = {8..15, 24..31}.
__device__ __forceinline__ v16h load_tileA(const _Float16* __restrict__ p,
                                           int r0, int c0, int ld) {
  int lane = threadIdx.x & 31;
  int row  = r0 + (lane & 15);
  int off  = c0 + ((lane >> 4) << 3);     // +0 or +8
  const _Float16* base = p + (size_t)row * ld + off;
  v8h lo = *(const v8h*)(base);           // K = off .. off+7
  v8h hi = *(const v8h*)(base + 16);      // K = off+16 .. off+23
  v16h out;
  #pragma unroll
  for (int i = 0; i < 8; ++i) { out[i] = lo[i]; out[8 + i] = hi[i]; }
  return out;
}

// B operand (32 contraction x 16 cols): element (K,N) = p[r0+N][c0+K]
// ISA layout (16-bit B): lane<16: N=lane, K = {0..7, 16..23}; lane>=16: N=lane-16, K = {8..15, 24..31}.
__device__ __forceinline__ v16h load_tileB(const _Float16* __restrict__ p,
                                           int r0, int c0, int ld) {
  int lane = threadIdx.x & 31;
  int row  = r0 + (lane & 15);
  int off  = c0 + ((lane >> 4) << 3);     // +0 or +8
  const _Float16* base = p + (size_t)row * ld + off;
  v8h lo = *(const v8h*)(base);           // K = off .. off+7
  v8h hi = *(const v8h*)(base + 16);      // K = off+16 .. off+23
  v16h out;
  #pragma unroll
  for (int i = 0; i < 8; ++i) { out[i] = lo[i]; out[8 + i] = hi[i]; }
  return out;
}

// ---------------------------------------------------------------------------
// fp32 -> fp16 converters
// ---------------------------------------------------------------------------
__global__ void cvt_kernel(const float* __restrict__ in,
                           _Float16* __restrict__ out, int n) {
  int i = blockIdx.x * blockDim.x + threadIdx.x;
  if (i < n) out[i] = (_Float16)in[i];
}

// in: [rows][cols] fp32 -> out: [cols][rows] fp16
__global__ void cvt_transpose_kernel(const float* __restrict__ in,
                                     _Float16* __restrict__ out,
                                     int rows, int cols) {
  int i = blockIdx.x * blockDim.x + threadIdx.x;
  if (i < rows * cols) {
    int r = i / cols, c = i % cols;
    out[(size_t)c * rows + r] = (_Float16)in[i];
  }
}

// ---------------------------------------------------------------------------
// GEMM1: register-blocked 32x64 per wave.
// qkv[m][n] = Xh[m][:] . WqkvT[n][:] + b ; scatter to Qh/Kh/Vt (fp16)
// Qh,Kh: [b][h][t][d]   Vt: [b][h][d][t]
// ---------------------------------------------------------------------------
__global__ void gemm_qkv_kernel(const _Float16* __restrict__ Xh,
                                const _Float16* __restrict__ WqkvT,
                                const float* __restrict__ bqkv,
                                _Float16* __restrict__ Qh,
                                _Float16* __restrict__ Kh,
                                _Float16* __restrict__ Vt) {
  int lane  = threadIdx.x & 31;
  int wave  = threadIdx.x >> 5;
  int mbase = blockIdx.x << 5;                        // 32 rows
  int nbase = ((blockIdx.y << 2) + wave) << 6;        // 64 cols
  v8f acc[2][4];
  #pragma unroll
  for (int i = 0; i < 2; ++i)
    #pragma unroll
    for (int j = 0; j < 4; ++j) acc[i][j] = v8f{};

  for (int k = 0; k < CC; k += 32) {
    v16h a0 = load_tileA(Xh, mbase,      k, CC);
    v16h a1 = load_tileA(Xh, mbase + 16, k, CC);
    v16h b0 = load_tileB(WqkvT, nbase,      k, CC);
    v16h b1 = load_tileB(WqkvT, nbase + 16, k, CC);
    v16h b2 = load_tileB(WqkvT, nbase + 32, k, CC);
    v16h b3 = load_tileB(WqkvT, nbase + 48, k, CC);
    acc[0][0] = wmma32(a0, b0, acc[0][0]);
    acc[0][1] = wmma32(a0, b1, acc[0][1]);
    acc[0][2] = wmma32(a0, b2, acc[0][2]);
    acc[0][3] = wmma32(a0, b3, acc[0][3]);
    acc[1][0] = wmma32(a1, b0, acc[1][0]);
    acc[1][1] = wmma32(a1, b1, acc[1][1]);
    acc[1][2] = wmma32(a1, b2, acc[1][2]);
    acc[1][3] = wmma32(a1, b3, acc[1][3]);
  }

  int srcoff = (lane >> 4) << 3;
  #pragma unroll
  for (int j = 0; j < 4; ++j) {
    int n = nbase + (j << 4) + (lane & 15);
    float bias = bqkv[n];
    #pragma unroll
    for (int i = 0; i < 2; ++i) {
      #pragma unroll
      for (int r = 0; r < 8; ++r) {
        int mrow = mbase + (i << 4) + r + srcoff;
        int bb = mrow >> 11;            // / T
        int t  = mrow & (TT - 1);
        _Float16 hv = (_Float16)(acc[i][j][r] + bias);
        if (n < CC) {
          int h = n >> 6, d = n & 63;
          Qh[(((size_t)(bb * HH + h)) * TT + t) * HD + d] = hv;
        } else if (n < 2 * CC) {
          int n2 = n - CC; int h = n2 >> 6, d = n2 & 63;
          Kh[(((size_t)(bb * HH + h)) * TT + t) * HD + d] = hv;
        } else {
          int n2 = n - 2 * CC; int h = n2 >> 6, d = n2 & 63;
          Vt[(((size_t)(bb * HH + h)) * HD + d) * TT + t] = hv;  // transposed
        }
      }
    }
  }
}

// ---------------------------------------------------------------------------
// Flash attention: one wave per (b,h,32-query group) = 2 q-tiles sharing all
// K/V tile loads. Scores computed transposed (St = Ktile x Q^T) so St's
// D-layout is exactly P's A-operand layout (no cross-lane repack).
// ---------------------------------------------------------------------------
__global__ void attn_kernel(const _Float16* __restrict__ Qh,
                            const _Float16* __restrict__ Kh,
                            const _Float16* __restrict__ Vt,
                            _Float16* __restrict__ Att) {
  int lane  = threadIdx.x & 31;
  int w     = blockIdx.x * (blockDim.x >> 5) + (threadIdx.x >> 5);
  int qg    = w & (TT / 32 - 1);            // 64 query groups per (b,h)
  int bh    = w >> 6;
  int h     = bh & (HH - 1);
  int b     = bh >> 4;
  int qbase = qg << 5;                      // 32 queries
  const _Float16* Qp = Qh + (size_t)bh * TT * HD;
  const _Float16* Kp = Kh + (size_t)bh * TT * HD;
  const _Float16* Vp = Vt + (size_t)bh * HD * TT;

  // Q^T as B operand: [q-tile][d-chunk]
  v16h qb[2][2];
  #pragma unroll
  for (int qt = 0; qt < 2; ++qt) {
    qb[qt][0] = load_tileB(Qp, qbase + (qt << 4), 0,  HD);
    qb[qt][1] = load_tileB(Qp, qbase + (qt << 4), 32, HD);
  }

  v8f O[2][4];                              // [q-tile][d-tile]
  #pragma unroll
  for (int qt = 0; qt < 2; ++qt)
    #pragma unroll
    for (int t = 0; t < 4; ++t) O[qt][t] = v8f{};
  float m[2] = {-1e30f, -1e30f};
  float l[2] = {0.0f, 0.0f};
  int srcoff = (lane >> 4) << 3;            // 0 or 8 (lane-half M offset)
  int kmax   = qbase + 32;

  for (int kb = 0; kb < kmax; kb += 32) {
    // shared K tiles: [k-half][d-chunk]
    v16h ka[2][2];
    ka[0][0] = load_tileA(Kp, kb,      0,  HD);
    ka[0][1] = load_tileA(Kp, kb,      32, HD);
    ka[1][0] = load_tileA(Kp, kb + 16, 0,  HD);
    ka[1][1] = load_tileA(Kp, kb + 16, 32, HD);

    v16h pA[2];
    #pragma unroll
    for (int qt = 0; qt < 2; ++qt) {
      int q = qbase + (qt << 4) + (lane & 15);
      v8f s0 = {}, s1 = {};
      s0 = wmma32(ka[0][0], qb[qt][0], s0);
      s0 = wmma32(ka[0][1], qb[qt][1], s0);
      s1 = wmma32(ka[1][0], qb[qt][0], s1);
      s1 = wmma32(ka[1][1], qb[qt][1], s1);

      float p0[8], p1[8];
      float pm = -1e30f;
      #pragma unroll
      for (int r = 0; r < 8; ++r) {
        int k0 = kb + srcoff + r;
        int k1 = k0 + 16;
        float a0 = (k0 <= q) ? s0[r] * ATT_SCALE : -1e30f;
        float a1 = (k1 <= q) ? s1[r] * ATT_SCALE : -1e30f;
        p0[r] = a0; p1[r] = a1;
        pm = fmaxf(pm, fmaxf(a0, a1));
      }
      pm = fmaxf(pm, __shfl_xor(pm, 16, 32));  // combine lane halves (same q)
      float mn = fmaxf(m[qt], pm);
      float alpha = __expf(m[qt] - mn);
      float psum = 0.0f;
      #pragma unroll
      for (int r = 0; r < 8; ++r) {
        p0[r] = __expf(p0[r] - mn);
        p1[r] = __expf(p1[r] - mn);
        psum += p0[r] + p1[r];
      }
      psum += __shfl_xor(psum, 16, 32);
      l[qt] = l[qt] * alpha + psum;
      m[qt] = mn;

      // rescale O rows: row r of O is q-row r+srcoff; alpha lives in lane q
      float ar[8];
      #pragma unroll
      for (int r = 0; r < 8; ++r) ar[r] = __shfl(alpha, r + srcoff, 32);
      #pragma unroll
      for (int t = 0; t < 4; ++t) {
        #pragma unroll
        for (int r = 0; r < 8; ++r) O[qt][t][r] *= ar[r];
      }

      // pack P (A operand, q rows x 32 k) straight from St D-layout
      #pragma unroll
      for (int r = 0; r < 8; ++r) {
        pA[qt][r]     = (_Float16)p0[r];   // K = kb + {0..7 | 8..15}
        pA[qt][8 + r] = (_Float16)p1[r];   // K = kb + {16..23 | 24..31}
      }
    }

    // O[q][d] += P(16x32k) x V(32k x 16d); shared V tiles, V^T contiguous in k
    #pragma unroll
    for (int t = 0; t < 4; ++t) {
      v16h vb = load_tileB(Vp, t * 16, kb, TT);
      O[0][t] = wmma32(pA[0], vb, O[0][t]);
      O[1][t] = wmma32(pA[1], vb, O[1][t]);
    }
  }

  int dcol = lane & 15;
  #pragma unroll
  for (int qt = 0; qt < 2; ++qt) {
    float linv = 1.0f / l[qt];
    float lr[8];
    #pragma unroll
    for (int r = 0; r < 8; ++r) lr[r] = __shfl(linv, r + srcoff, 32);
    #pragma unroll
    for (int t = 0; t < 4; ++t) {
      #pragma unroll
      for (int r = 0; r < 8; ++r) {
        int trow = qbase + (qt << 4) + r + srcoff;
        size_t idx = ((size_t)(b * TT + trow)) * CC + h * HD + t * 16 + dcol;
        Att[idx] = (_Float16)(O[qt][t][r] * lr[r]);
      }
    }
  }
}

// ---------------------------------------------------------------------------
// GEMM2: register-blocked 32x64 per wave.
// out[m][n] = Att[m][:] . WoutT[n][:] + b_out   (fp32 output)
// ---------------------------------------------------------------------------
__global__ void gemm_out_kernel(const _Float16* __restrict__ Att,
                                const _Float16* __restrict__ WoutT,
                                const float* __restrict__ bout,
                                float* __restrict__ out) {
  int lane  = threadIdx.x & 31;
  int wave  = threadIdx.x >> 5;
  int mbase = blockIdx.x << 5;
  int nbase = ((blockIdx.y << 2) + wave) << 6;
  v8f acc[2][4];
  #pragma unroll
  for (int i = 0; i < 2; ++i)
    #pragma unroll
    for (int j = 0; j < 4; ++j) acc[i][j] = v8f{};

  for (int k = 0; k < CC; k += 32) {
    v16h a0 = load_tileA(Att, mbase,      k, CC);
    v16h a1 = load_tileA(Att, mbase + 16, k, CC);
    v16h b0 = load_tileB(WoutT, nbase,      k, CC);
    v16h b1 = load_tileB(WoutT, nbase + 16, k, CC);
    v16h b2 = load_tileB(WoutT, nbase + 32, k, CC);
    v16h b3 = load_tileB(WoutT, nbase + 48, k, CC);
    acc[0][0] = wmma32(a0, b0, acc[0][0]);
    acc[0][1] = wmma32(a0, b1, acc[0][1]);
    acc[0][2] = wmma32(a0, b2, acc[0][2]);
    acc[0][3] = wmma32(a0, b3, acc[0][3]);
    acc[1][0] = wmma32(a1, b0, acc[1][0]);
    acc[1][1] = wmma32(a1, b1, acc[1][1]);
    acc[1][2] = wmma32(a1, b2, acc[1][2]);
    acc[1][3] = wmma32(a1, b3, acc[1][3]);
  }

  int srcoff = (lane >> 4) << 3;
  #pragma unroll
  for (int j = 0; j < 4; ++j) {
    int n = nbase + (j << 4) + (lane & 15);
    float bias = bout[n];
    #pragma unroll
    for (int i = 0; i < 2; ++i) {
      #pragma unroll
      for (int r = 0; r < 8; ++r) {
        int mrow = mbase + (i << 4) + r + srcoff;
        out[(size_t)mrow * CC + n] = acc[i][j][r] + bias;
      }
    }
  }
}

// ---------------------------------------------------------------------------
extern "C" void kernel_launch(void* const* d_in, const int* in_sizes, int n_in,
                              void* d_out, int out_size, void* d_ws, size_t ws_size,
                              hipStream_t stream) {
  (void)in_sizes; (void)n_in; (void)out_size; (void)ws_size;
  const float* x     = (const float*)d_in[0];
  const float* w_qkv = (const float*)d_in[1];
  const float* b_qkv = (const float*)d_in[2];
  const float* w_out = (const float*)d_in[3];
  const float* b_out = (const float*)d_in[4];
  float* out = (float*)d_out;

  char* ws = (char*)d_ws;                       // needs 40 MB total
  _Float16* Xh    = (_Float16*)(ws);                         // 8 MB
  _Float16* WqkvT = (_Float16*)(ws + ((size_t)8  << 20));    // 6 MB
  _Float16* WoutT = (_Float16*)(ws + ((size_t)14 << 20));    // 2 MB
  _Float16* Qh    = (_Float16*)(ws + ((size_t)16 << 20));    // 8 MB
  _Float16* Kh    = (_Float16*)(ws + ((size_t)24 << 20));    // 8 MB
  _Float16* Vt    = (_Float16*)(ws + ((size_t)32 << 20));    // 8 MB
  _Float16* Att   = Xh;   // Xh dead after gemm_qkv; reuse for attention output

  cvt_kernel<<<(MROWS * CC) / 256, 256, 0, stream>>>(x, Xh, MROWS * CC);
  cvt_transpose_kernel<<<(CC * NQKV) / 256, 256, 0, stream>>>(w_qkv, WqkvT, CC, NQKV);
  cvt_transpose_kernel<<<(CC * CC) / 256, 256, 0, stream>>>(w_out, WoutT, CC, CC);

  gemm_qkv_kernel<<<dim3(MROWS / 32, NQKV / 64 / 4), 128, 0, stream>>>(
      Xh, WqkvT, b_qkv, Qh, Kh, Vt);

  attn_kernel<<<(BB * HH * (TT / 32)) / 4, 128, 0, stream>>>(Qh, Kh, Vt, Att);

  gemm_out_kernel<<<dim3(MROWS / 32, CC / 64 / 4), 128, 0, stream>>>(
      Att, WoutT, b_out, out);
}
